// AttentionBlock_57080115364858
// MI455X (gfx1250) — matrix operand order repaired and checked by reading, not compiled, not run on hardware
//
#include <hip/hip_runtime.h>
#include <hip/hip_bf16.h>

// ---------------------------------------------------------------------------
// AttentionBlock for MI455X (gfx1250, wave32, WMMA).
// f16 inputs to v_wmma_f32_16x16x32_f16, fp32 accumulate,
// fp32 RMSNorm / softmax / RoPE / residual.
//
// Workspace (~150.6 MB):
//   [OFF_WT ] f16 weights, transposed [mat][col][k], 4*384*384*2
//   [OFF_Q  ] f16 Q  [b][h][n(384 pad)][32]
//   [OFF_K  ] f16 K  [b][h][m(384 pad)][32]
//   [OFF_VT ] f16 V^T[b][h][d=32][m(384 pad)]
//   [OFF_O  ] f16 attn out [b][n(368 pad)][384]
// ---------------------------------------------------------------------------

typedef _Float16 v16h __attribute__((ext_vector_type(16)));
typedef float    v8f  __attribute__((ext_vector_type(8)));

#define DIMC      384
#define NHEADS    12
#define HD        32
#define NPTS      361
#define NT        23            // ceil(361/16)
#define NROWPAD   368           // 23*16
#define MPAD      384           // 12 m-chunks of 32
#define KCH       12            // 384/32 k-chunks

static const size_t OFF_WT = 0;
static const size_t SZ_WT  = (size_t)4 * 384 * 384 * 2;
static const size_t OFF_Q  = OFF_WT + SZ_WT;
static const size_t SZ_QKV = (size_t)128 * 12 * 384 * 32 * 2;
static const size_t OFF_K  = OFF_Q + SZ_QKV;
static const size_t OFF_VT = OFF_K + SZ_QKV;
static const size_t OFF_O  = OFF_VT + SZ_QKV;

// --- fragment load helpers (16B-aligned b128 pairs) -------------------------
__device__ __forceinline__ v16h ld16h(const void* p) {                 // 32B contiguous
    union { uint4 q[2]; v16h h; } u;
    u.q[0] = *(const uint4*)p;
    u.q[1] = *(const uint4*)((const char*)p + 16);
    return u.h;
}
__device__ __forceinline__ v16h ld16h_split(const void* p0, const void* p1) {
    union { uint4 q[2]; v16h h; } u;
    u.q[0] = *(const uint4*)p0;
    u.q[1] = *(const uint4*)p1;
    return u.h;
}
__device__ __forceinline__ v8f wmmaf(v16h a, v16h b, v8f c) {
    return __builtin_amdgcn_wmma_f32_16x16x32_f16(false, a, false, b, (short)0, c,
                                                  false, false);
}

// ===========================================================================
// Kernel 0: weights f32[k][col] -> f16 transposed [mat][col][k]
// ===========================================================================
__global__ void k_wprep(const float* __restrict__ qw, const float* __restrict__ kw,
                        const float* __restrict__ vw, const float* __restrict__ ow,
                        _Float16* __restrict__ wt) {
    int i = blockIdx.x * 256 + threadIdx.x;
    if (i >= 4 * 384 * 384) return;
    int w   = i / (384 * 384);
    int r   = i - w * (384 * 384);
    int col = r / 384;
    int k   = r - col * 384;
    const float* src = (w == 0) ? qw : (w == 1) ? kw : (w == 2) ? vw : ow;
    wt[i] = (_Float16)src[k * 384 + col];
}

// --- QKV tile epilogue: RoPE (Q,K) + scatter store --------------------------
__device__ __forceinline__ void qkv_epilogue(v8f acc, int mat, int c0, int n0, int b,
                                             int lane, const float* __restrict__ cost,
                                             const float* __restrict__ sint,
                                             _Float16* __restrict__ wQ,
                                             _Float16* __restrict__ wK,
                                             _Float16* __restrict__ wVt) {
    const int lh = lane & 15;
    const int h  = (c0 + lh) >> 5;
    const int d  = (c0 + lh) & 31;
    if (mat < 2) {  // RoPE: pair partner sits on the adjacent lane in C layout
        const bool evenc = ((d & 1) == 0);
#pragma unroll
        for (int v = 0; v < 8; ++v) {
            int   n   = n0 + v + ((lane < 16) ? 0 : 8);
            float val = acc[v];
            float pr  = __shfl_xor(val, 1);
            float rot = evenc ? -pr : pr;
            float cs = 1.f, sn = 0.f;
            if (n < NPTS) { cs = cost[n * HD + d]; sn = sint[n * HD + d]; }
            acc[v] = val * cs + rot * sn;
        }
    }
#pragma unroll
    for (int v = 0; v < 8; ++v) {
        int n = n0 + v + ((lane < 16) ? 0 : 8);
        _Float16 hv = (_Float16)acc[v];
        size_t idx = (((size_t)b * NHEADS + h) * MPAD + n) * HD + d;
        if (mat == 0)      wQ[idx] = hv;
        else if (mat == 1) wK[idx] = hv;
        else wVt[(((size_t)b * NHEADS + h) * HD + d) * MPAD + n] = hv;
    }
}

// ===========================================================================
// Kernel 1: RMSNorm + QKV projection (WMMA) + RoPE.
// grid (23, 128), block 768 (24 waves). Wave owns exactly 3 of 72 tiles:
// no outer tile loop => nothing for LICM to hoist => no spills.
// ===========================================================================
__global__ __launch_bounds__(768) void
k_qkv(const float* __restrict__ x, const float* __restrict__ normw,
      const float* __restrict__ cost, const float* __restrict__ sint,
      const _Float16* __restrict__ wt,
      _Float16* __restrict__ wQ, _Float16* __restrict__ wK,
      _Float16* __restrict__ wVt) {
    __shared__ float xf[16 * 385];
    __shared__ float part[768];
    __shared__ float sc[16];
    __shared__ alignas(16) _Float16 xnh[16 * 384];

    const int tid = threadIdx.x;
    const int nt  = blockIdx.x;
    const int b   = blockIdx.y;
    const int n0  = nt * 16;

    // ---- load x tile (16 x 384), zero-pad rows >= 361 ----------------------
    for (int i = tid; i < 16 * 384; i += 768) {
        int r = i & 15, c = i >> 4;
        int n = n0 + r;
        xf[r * 385 + c] = (n < NPTS) ? x[((size_t)b * DIMC + c) * NPTS + n] : 0.f;
    }
    __syncthreads();
    // ---- RMSNorm: 48 partial-sum threads per row ---------------------------
    {
        int r = tid & 15, s = tid >> 4;   // s in 0..47
        float acc = 0.f;
        for (int c = s; c < DIMC; c += 48) { float v = xf[r * 385 + c]; acc += v * v; }
        part[tid] = acc;
    }
    __syncthreads();
    if (tid < 16) {
        float s = 0.f;
        for (int j = 0; j < 48; ++j) s += part[tid + 16 * j];
        sc[tid] = rsqrtf(s * (1.0f / DIMC) + 1e-6f);
    }
    __syncthreads();
    for (int i = tid; i < 16 * 384; i += 768) {
        int r = i & 15, c = i >> 4;
        xnh[r * 384 + c] = (_Float16)(xf[r * 385 + c] * normw[c] * sc[r]);
    }
    __syncthreads();

    // ---- QKV via WMMA: wave w -> tiles 3w..3w+2 ----------------------------
    const int wave = tid >> 5, lane = tid & 31;
    const int lh   = lane & 15;
    const int selA = (lane < 16) ? 0 : 32;   // bytes, A-fragment half-select
    const int selH = (lane < 16) ? 0 : 16;   // halves, B-fragment K half-select
    const int tbase = wave * 3;

    int matU[3], c0U[3];
    const _Float16* bpU[3];
#pragma unroll
    for (int u = 0; u < 3; ++u) {
        int t   = tbase + u;
        matU[u] = t / 24;
        c0U[u]  = (t - matU[u] * 24) * 16;
        bpU[u]  = wt + (size_t)matU[u] * 384 * 384 + (size_t)(c0U[u] + lh) * 384 + selH;
    }
    v8f zz = {};
    v8f accU[3];
#pragma unroll
    for (int u = 0; u < 3; ++u) accU[u] = zz;

#pragma unroll
    for (int kc = 0; kc < KCH; ++kc) {
        const int k0 = kc * 32;
        const char* xrow = (const char*)xnh + (size_t)lh * 768 + (size_t)k0 * 2;
        v16h a = ld16h_split(xrow + selA, xrow + 16 + selA);
#pragma unroll
        for (int u = 0; u < 3; ++u)
            accU[u] = wmmaf(a, ld16h(bpU[u] + k0), accU[u]);
    }
#pragma unroll
    for (int u = 0; u < 3; ++u)
        qkv_epilogue(accU[u], matU[u], c0U[u], n0, b, lane, cost, sint, wQ, wK, wVt);

    // zero-fill pad rows m=368..383 of K and V^T (masked later; must be non-NaN)
    if (nt == NT - 1) {
        for (int i = tid; i < NHEADS * 16 * 32; i += 768) {
            int h = i / (16 * 32);
            int r = (i >> 5) & 15;
            int d = i & 31;
            int m = NROWPAD + r;
            wK [(((size_t)b * NHEADS + h) * MPAD + m) * HD + d]  = (_Float16)0.f;
            wVt[(((size_t)b * NHEADS + h) * HD + d) * MPAD + m] = (_Float16)0.f;
        }
    }
}

// ===========================================================================
// Kernel 2: flash attention, one wave per (b, head, 16-row n-tile).
// grid (23, 128), block 384 (12 waves, wave = head).
// ===========================================================================
__global__ __launch_bounds__(384) void
k_attn(const _Float16* __restrict__ wQ, const _Float16* __restrict__ wK,
       const _Float16* __restrict__ wVt, _Float16* __restrict__ wO) {
    __shared__ alignas(16) _Float16 pb[12][16 * 32];   // per-wave P transpose buffer

    const int lane = threadIdx.x & 31;
    const int h    = threadIdx.x >> 5;       // wave == head
    const int nt   = blockIdx.x;
    const int b    = blockIdx.y;
    const int n0   = nt * 16;

    const int lh   = lane & 15;
    const int selA = (lane < 16) ? 0 : 32;   // bytes (A frag)
    const int selB = (lane < 16) ? 0 : 32;   // bytes (K B frag)
    const int selM = (lane < 16) ? 0 : 16;   // halves (V^T B frag)

    const char* qb = (const char*)(wQ  + (((size_t)b * NHEADS + h) * MPAD) * HD);
    const char* kb = (const char*)(wK  + (((size_t)b * NHEADS + h) * MPAD) * HD);
    const char* vb = (const char*)(wVt + (((size_t)b * NHEADS + h) * HD) * MPAD);

    // Q A-fragment held in registers for the whole loop
    const char* qrow = qb + (size_t)(n0 + lh) * 64;
    v16h qa = ld16h_split(qrow + selA, qrow + 16 + selA);

    v8f o0 = {}, o1 = {};
    float mrow[8], lrow[8];
#pragma unroll
    for (int v = 0; v < 8; ++v) { mrow[v] = -3e38f; lrow[v] = 0.f; }

    const float scale = 0.17677669529663687f;  // 1/sqrt(32)
    _Float16* pbw = &pb[h][0];

    for (int mc = 0; mc < MPAD / 32; ++mc) {
        const int m0 = mc * 32;
        v16h kb0 = ld16h(kb + (size_t)(m0 + lh) * 64 + selB);
        v16h kb1 = ld16h(kb + (size_t)(m0 + 16 + lh) * 64 + selB);
        if (mc + 1 < MPAD / 32)
            __builtin_prefetch(kb + (size_t)(m0 + 32 + lh) * 64, 0, 0);
        v8f z = {};
        v8f s0 = wmmaf(qa, kb0, z);
        v8f s1 = wmmaf(qa, kb1, z);

        const bool v0ok = (m0 + lh)      < NPTS;
        const bool v1ok = (m0 + 16 + lh) < NPTS;
#pragma unroll
        for (int v = 0; v < 8; ++v) {
            float a = v0ok ? s0[v] * scale : -3e38f;
            float c = v1ok ? s1[v] * scale : -3e38f;
            float t = fmaxf(a, c);
            t = fmaxf(t, __shfl_xor(t, 1));
            t = fmaxf(t, __shfl_xor(t, 2));
            t = fmaxf(t, __shfl_xor(t, 4));
            t = fmaxf(t, __shfl_xor(t, 8));
            float mnew  = fmaxf(mrow[v], t);
            float alpha = __expf(mrow[v] - mnew);
            mrow[v] = mnew;
            float p0 = __expf(a - mnew);
            float p1 = __expf(c - mnew);
            float rs = p0 + p1;
            rs += __shfl_xor(rs, 1);
            rs += __shfl_xor(rs, 2);
            rs += __shfl_xor(rs, 4);
            rs += __shfl_xor(rs, 8);
            lrow[v] = lrow[v] * alpha + rs;
            o0[v] *= alpha;
            o1[v] *= alpha;
            int row = v + ((lane < 16) ? 0 : 8);
            pbw[row * 32 + lh]      = (_Float16)p0;   // C layout -> row-major LDS
            pbw[row * 32 + 16 + lh] = (_Float16)p1;
        }
        // wait for this wave's LDS stores before the transposed read-back
        asm volatile("s_wait_dscnt 0" ::: "memory");

        const char* prow = (const char*)pbw + (size_t)lh * 64;
        v16h pa  = ld16h_split(prow + selA, prow + 16 + selA);
        v16h vb0 = ld16h((const char*)vb + (size_t)lh * 768        + (size_t)(m0 + selM) * 2);
        v16h vb1 = ld16h((const char*)vb + (size_t)(16 + lh) * 768 + (size_t)(m0 + selM) * 2);
        o0 = wmmaf(pa, vb0, o0);
        o1 = wmmaf(pa, vb1, o1);
        // LDS ops are in-order within a wave; fence keeps compiler honest
        asm volatile("" ::: "memory");
    }

    _Float16* ob = wO + (size_t)b * NROWPAD * DIMC;
#pragma unroll
    for (int v = 0; v < 8; ++v) {
        float inv = 1.f / lrow[v];
        int   n   = n0 + v + ((lane < 16) ? 0 : 8);
        ob[(size_t)n * DIMC + h * HD + lh]      = (_Float16)(o0[v] * inv);
        ob[(size_t)n * DIMC + h * HD + 16 + lh] = (_Float16)(o1[v] * inv);
    }
}

// ===========================================================================
// Kernel 3: out projection (WMMA) + residual, fp32 output (B, 361, 384).
// grid (23, 128), block 256 (8 waves x 3 tiles, single group => no spills).
// ===========================================================================
__global__ __launch_bounds__(256) void
k_out(const _Float16* __restrict__ wO, const _Float16* __restrict__ wto,
      const float* __restrict__ x, float* __restrict__ out) {
    const int lane = threadIdx.x & 31;
    const int wave = threadIdx.x >> 5;
    const int nt   = blockIdx.x;
    const int b    = blockIdx.y;
    const int n0   = nt * 16;

    const int lh   = lane & 15;
    const int selA = (lane < 16) ? 0 : 32;   // bytes
    const int selH = (lane < 16) ? 0 : 16;   // halves

    const char* ob = (const char*)(wO + (size_t)b * NROWPAD * DIMC);
    const int tbase = wave * 3;

    const _Float16* bpU[3];
    int c0U[3];
#pragma unroll
    for (int u = 0; u < 3; ++u) {
        c0U[u] = (tbase + u) * 16;
        bpU[u] = wto + (size_t)(c0U[u] + lh) * 384 + selH;
    }
    v8f zz = {};
    v8f accU[3];
#pragma unroll
    for (int u = 0; u < 3; ++u) accU[u] = zz;

#pragma unroll
    for (int kc = 0; kc < KCH; ++kc) {
        const int k0 = kc * 32;
        const char* orow = ob + (size_t)(n0 + lh) * 768 + (size_t)k0 * 2;
        v16h a = ld16h_split(orow + selA, orow + 16 + selA);
#pragma unroll
        for (int u = 0; u < 3; ++u)
            accU[u] = wmmaf(a, ld16h(bpU[u] + k0), accU[u]);
    }
#pragma unroll
    for (int u = 0; u < 3; ++u) {
        const int c = c0U[u] + lh;
#pragma unroll
        for (int v = 0; v < 8; ++v) {
            int n = n0 + v + ((lane < 16) ? 0 : 8);
            if (n < NPTS) {
                float res = x[((size_t)b * DIMC + c) * NPTS + n];
                out[((size_t)b * NPTS + n) * DIMC + c] = res + accU[u][v];
            }
        }
    }
}

// ===========================================================================
extern "C" void kernel_launch(void* const* d_in, const int* in_sizes, int n_in,
                              void* d_out, int out_size, void* d_ws, size_t ws_size,
                              hipStream_t stream) {
    const float* x     = (const float*)d_in[0];
    const float* qw    = (const float*)d_in[1];
    const float* kw    = (const float*)d_in[2];
    const float* vw    = (const float*)d_in[3];
    const float* ow    = (const float*)d_in[4];
    const float* normw = (const float*)d_in[5];
    const float* cost  = (const float*)d_in[6];
    const float* sint  = (const float*)d_in[7];
    float* out = (float*)d_out;

    char* ws = (char*)d_ws;
    _Float16* wt  = (_Float16*)(ws + OFF_WT);
    _Float16* wQ  = (_Float16*)(ws + OFF_Q);
    _Float16* wK  = (_Float16*)(ws + OFF_K);
    _Float16* wVt = (_Float16*)(ws + OFF_VT);
    _Float16* wOa = (_Float16*)(ws + OFF_O);

    k_wprep<<<(4 * 384 * 384 + 255) / 256, 256, 0, stream>>>(qw, kw, vw, ow, wt);
    k_qkv  <<<dim3(NT, 128), 768, 0, stream>>>(x, normw, cost, sint, wt, wQ, wK, wVt);
    k_attn <<<dim3(NT, 128), 384, 0, stream>>>(wQ, wK, wVt, wOa);
    k_out  <<<dim3(NT, 128), 256, 0, stream>>>(wOa, wt + (size_t)3 * 384 * 384, x, out);
}
